// PFAModel_44779329028254
// MI455X (gfx1250) — matrix operand order repaired and checked by reading, not compiled, not run on hardware
//
#include <hip/hip_runtime.h>
#include <hip/hip_bf16.h>
#include <math.h>

// Problem constants (match reference): B=1024, T=512, Q=128, S=4
#define PFA_B 1024
#define PFA_T 512
#define PFA_Q 128
#define PFA_S 4

typedef __attribute__((ext_vector_type(16))) _Float16 v16h;
typedef __attribute__((ext_vector_type(8)))  _Float16 v8h;
typedef __attribute__((ext_vector_type(8)))  float    v8f;

static __device__ inline v16h join16(v8h lo, v8h hi) {
    v16h r;
#pragma unroll
    for (int i = 0; i < 8; ++i) { r[i] = lo[i]; r[i + 8] = hi[i]; }
    return r;
}

// ---------------------------------------------------------------------------
// Prep: Pt[s][qout][qin] = softmax(T_logits[qin, s, :])[qout], f16,
// qin contiguous.  This serves directly as the WMMA A-matrix (P_s^T):
// row M = qout, K = qin.  One block per (qin, s) row; 128-wide softmax.
// ---------------------------------------------------------------------------
__global__ __launch_bounds__(128) void pfa_prep_kernel(const float* __restrict__ Tl,
                                                       _Float16* __restrict__ Pt) {
    const int row = blockIdx.x;        // row = qin * 4 + s
    const int q   = threadIdx.x;       // destination state qout
    __shared__ float red[128];

    float v = Tl[row * PFA_Q + q];
    red[q] = v;
    __syncthreads();
#pragma unroll
    for (int off = 64; off > 0; off >>= 1) {
        if (q < off) red[q] = fmaxf(red[q], red[q + off]);
        __syncthreads();
    }
    float m = red[0];
    __syncthreads();
    float e = expf(v - m);
    red[q] = e;
    __syncthreads();
#pragma unroll
    for (int off = 64; off > 0; off >>= 1) {
        if (q < off) red[q] += red[q + off];
        __syncthreads();
    }
    float sum = red[0];

    const int qin = row >> 2;
    const int s   = row & 3;
    Pt[(s * PFA_Q + q) * PFA_Q + qin] = (_Float16)(e / sum);
}

// ---------------------------------------------------------------------------
// Main forward kernel: per step computes Y^T = P_s^T @ W^T per (s, mtile) wave.
// Grid: B/16 = 64 blocks, 1024 threads (32 waves).
// Wave wid = s*8 + mtile: symbol s, output states [mtile*16, mtile*16+16).
// C layout puts one batch row per lane -> per-lane-uniform symbol/length
// predicate -> single predicated ds_store_b128 commits the update.
// ---------------------------------------------------------------------------
#define W_STRIDE 136   // 128 + 8 halves padding -> conflict-free ds_load_b128

__global__ __launch_bounds__(1024, 1) void pfa_forward_kernel(
        const int*      __restrict__ x,        // [B, T]
        const int*      __restrict__ lengths,  // [B]
        const _Float16* __restrict__ Pt,       // [S][qout][qin] f16
        const float*    __restrict__ f_logits, // [Q]
        float*          __restrict__ out) {    // [B]

    __shared__ __align__(16) _Float16      wT[16 * W_STRIDE];   // [batch][state]
    __shared__ __align__(16) unsigned char xsymT[PFA_T * 16];   // [t][batch]

    const int tid    = threadIdx.x;
    const int lane   = tid & 31;
    const int wid    = tid >> 5;       // 0..31
    const int s      = wid >> 3;       // symbol this wave applies
    const int mtile  = wid & 7;        // 16-row output-state tile
    const int batch0 = blockIdx.x * 16;
    const int bl     = lane & 15;      // batch row held by this lane (N dim)
    const int hi     = lane >> 4;      // half-wave selector

    // ---- init: w[b][:] = one-hot(state 0) ---------------------------------
    for (int i = tid; i < 16 * W_STRIDE; i += 1024)
        wT[i] = (_Float16)((i % W_STRIDE) == 0 ? 1.0f : 0.0f);

    // ---- stage symbols transposed: xsymT[t*16 + b] ------------------------
    for (int i = tid; i < 16 * PFA_T; i += 1024) {
        const int m = i >> 9, t = i & (PFA_T - 1);
        xsymT[t * 16 + m] = (unsigned char)x[(batch0 + m) * PFA_T + t];
    }
    const int lenN = lengths[batch0 + bl];   // length of this lane's batch row

    // ---- A fragments: P_s^T rows [mtile*16, +16), K = 0..127, in VGPRs ----
    // A layout (16x32 f16): lane M = lane&15; elems 0..7 -> K=8*hi+e,
    // elems 8..15 -> K=16+8*hi+(e-8).
    v16h Af[4];
    {
        const int mrow = mtile * 16 + bl;
#pragma unroll
        for (int kb = 0; kb < 4; ++kb) {
            const _Float16* p = Pt + ((s * PFA_Q + mrow) * PFA_Q + kb * 32 + hi * 8);
            Af[kb] = join16(*(const v8h*)p, *(const v8h*)(p + 16));
        }
    }
    __syncthreads();

    // B layout (32x16 f16): lane N = bl; elems 0..7 -> K = 8*hi + e, elems 8..15 -> K = 16+8*hi+(e-8).
    const _Float16*      bptr = &wT[bl * W_STRIDE + hi * 8];
    _Float16*            sptr = &wT[bl * W_STRIDE + mtile * 16 + hi * 8];
    const unsigned char* symp = &xsymT[bl];

    for (int t = 0; t < PFA_T; ++t) {
        // ---- load W^T fragments (distinct regs -> staggered waits) --------
        v16h B0 = join16(*(const v8h*)(bptr +  0), *(const v8h*)(bptr +  16));
        v16h B1 = join16(*(const v8h*)(bptr + 32), *(const v8h*)(bptr +  48));
        v16h B2 = join16(*(const v8h*)(bptr + 64), *(const v8h*)(bptr +  80));
        v16h B3 = join16(*(const v8h*)(bptr + 96), *(const v8h*)(bptr + 112));
        const int sym = symp[t * 16];

        // ---- Y^T(tile) = P_s^T @ W^T, f32 accumulate ----------------------
        v8f acc = {};
        acc = __builtin_amdgcn_wmma_f32_16x16x32_f16(false, Af[0], false, B0,
                                                     (short)0, acc, false, false);
        acc = __builtin_amdgcn_wmma_f32_16x16x32_f16(false, Af[1], false, B1,
                                                     (short)0, acc, false, false);
        acc = __builtin_amdgcn_wmma_f32_16x16x32_f16(false, Af[2], false, B2,
                                                     (short)0, acc, false, false);
        acc = __builtin_amdgcn_wmma_f32_16x16x32_f16(false, Af[3], false, B3,
                                                     (short)0, acc, false, false);

        // C layout: reg r holds (state = mtile*16 + hi*8 + r, batch = bl):
        // the lane's 8 results are 8 contiguous states of ONE batch row.
        v8h yh;
#pragma unroll
        for (int r = 0; r < 8; ++r) yh[r] = (_Float16)acc[r];

        __syncthreads();   // all waves done reading old W^T

        // ---- commit: exactly one symbol-wave wins per (batch,row); rows
        // past their length are never written -> carry old alpha ------------
        if (sym == s && t < lenN)
            *(v8h*)sptr = yh;

        __syncthreads();   // new W^T visible before next step's loads
    }

    // ---- out[b] = log( sum_q w[b][q] * exp(f[q]) )  (sum_q w == 1) --------
    if (wid < 16) {
        const int M = wid;
        float sum = 0.0f;
#pragma unroll
        for (int kk = 0; kk < 4; ++kk) {
            const int k = lane + kk * 32;
            sum += (float)wT[M * W_STRIDE + k] * expf(f_logits[k]);
        }
#pragma unroll
        for (int off = 16; off > 0; off >>= 1)
            sum += __shfl_xor(sum, off, 32);
        if (lane == 0) out[batch0 + M] = logf(sum);
    }
}

// ---------------------------------------------------------------------------
extern "C" void kernel_launch(void* const* d_in, const int* in_sizes, int n_in,
                              void* d_out, int out_size, void* d_ws, size_t ws_size,
                              hipStream_t stream) {
    (void)in_sizes; (void)n_in; (void)out_size; (void)ws_size;

    const int*   x        = (const int*)d_in[0];
    const int*   lengths  = (const int*)d_in[1];
    const float* T_logits = (const float*)d_in[2];
    const float* f_logits = (const float*)d_in[3];
    float*       out      = (float*)d_out;

    _Float16* Pt = (_Float16*)d_ws;   // 4*128*128 f16 = 128 KB scratch

    pfa_prep_kernel<<<PFA_Q * PFA_S, PFA_Q, 0, stream>>>(T_logits, Pt);
    pfa_forward_kernel<<<PFA_B / 16, 1024, 0, stream>>>(x, lengths, Pt, f_logits, out);
}
